// MultiLevelSCSA_scaler_70093866270972
// MI455X (gfx1250) — matrix operand order repaired and checked by reading, not compiled, not run on hardware
//
#include <hip/hip_runtime.h>
#include <hip/hip_bf16.h>

typedef __bf16 bf16;
typedef bf16 bf16x8  __attribute__((ext_vector_type(8)));
typedef bf16 bf16x16 __attribute__((ext_vector_type(16)));
typedef float floatx8 __attribute__((ext_vector_type(8)));

#define NB 16
#define NC 256
#define NHW 64
#define NSP 4096          // 64*64
#define NM  65536         // B * N
#define NHEADS 4
#define NHD 64
#define NGROUPS 4
#define GN_COUNT 262144.0f   // (C/G)*H*W

// ---------------- workspace layout (bytes) ----------------
static constexpr size_t OFF_STATS = 0;                         //   512 f  (scale,b,group,{s,s2})
static constexpr size_t OFF_KSUM  = 2048;                      //  4096 f  (b, c)
static constexpr size_t OFF_KV    = 18432;                     // 262144 f (b*h, d, e)
static constexpr size_t ZERO_FLOATS = (18432 + 262144*4) / 4;  // stats+ksum+kv contiguous
static constexpr size_t OFF_Z     = 1067008;                   // 262144 f (m, head)
static constexpr size_t OFF_QKWB  = 2115584;                   // 131072 bf16
static constexpr size_t OFF_FUSED = 2377728;                   // 16.7M f  (b,c,h,w)
static constexpr size_t OFF_XS    = OFF_FUSED + 67108864;      // 16.7M bf16 (b,n,c)   == v
static constexpr size_t OFF_QROPE = OFF_XS    + 33554432;      // 16.7M bf16
static constexpr size_t OFF_KROPE = OFF_QROPE + 33554432;      // 16.7M bf16
static constexpr size_t OFF_QBUF  = OFF_KROPE + 33554432;      // 16.7M bf16 (q, pre-rope)
static constexpr size_t OFF_KBUF  = OFF_QBUF  + 33554432;      // 16.7M bf16 (k, pre-rope)
static constexpr size_t OFF_XO    = OFF_QBUF;                  // 16.7M f, aliases dead q/k bufs

// ---------------- CDNA5 async memory->LDS helpers (inline asm; ASYNCcnt tracked) -----
__device__ __forceinline__ void async_load_b128_to_lds(const void* gptr, void* lds_ptr) {
    // LDS aperture: flat addr low 32 bits == LDS byte offset (VDST payload).
    unsigned ldsa = (unsigned)(size_t)lds_ptr;
    unsigned long long ga = (unsigned long long)(size_t)gptr;
    asm volatile("global_load_async_to_lds_b128 %0, %1, off"
                 :: "v"(ldsa), "v"(ga) : "memory");
}
__device__ __forceinline__ void wait_asynccnt0() {
    asm volatile("s_wait_asynccnt 0x0" ::: "memory");
}

// ---------------- small utility kernels ----------------
__global__ void zero_f32(float* p, int n) {
    int i = blockIdx.x * blockDim.x + threadIdx.x;
    if (i < n) p[i] = 0.0f;
}

__global__ void cvt_bf16(const float* __restrict__ src, bf16* __restrict__ dst, int n) {
    int i = blockIdx.x * blockDim.x + threadIdx.x;
    if (i < n) dst[i] = (bf16)src[i];
}

// ---------------- fused depthwise conv helper (all 4 scales, each tap read once) -----
__device__ __forceinline__ void conv_all(const float* __restrict__ plane, int h, int w,
                                         const float* __restrict__ w3, const float* __restrict__ w5,
                                         const float* __restrict__ w7, const float* __restrict__ w9,
                                         float b3, float b5, float b7, float b9, float y[4]) {
    float a0 = b3, a1 = b5, a2 = b7, a3 = b9;
    #pragma unroll
    for (int dh = -4; dh <= 4; ++dh) {
        int hh = h + dh;
        bool hin = (hh >= 0) & (hh < NHW);
        #pragma unroll
        for (int dw = -4; dw <= 4; ++dw) {
            int ww = w + dw;
            float v = (hin && ww >= 0 && ww < NHW) ? plane[hh * NHW + ww] : 0.0f;
            a3 += v * w9[(dh + 4) * 9 + (dw + 4)];
            if (dh >= -3 && dh <= 3 && dw >= -3 && dw <= 3) a2 += v * w7[(dh + 3) * 7 + (dw + 3)];
            if (dh >= -2 && dh <= 2 && dw >= -2 && dw <= 2) a1 += v * w5[(dh + 2) * 5 + (dw + 2)];
            if (dh >= -1 && dh <= 1 && dw >= -1 && dw <= 1) a0 += v * w3[(dh + 1) * 3 + (dw + 1)];
        }
    }
    y[0] = a0; y[1] = a1; y[2] = a2; y[3] = a3;
}

// Pass 1: per-(b,c) plane, accumulate per-(scale,b,group) sum / sum^2 for GroupNorm
__global__ __launch_bounds__(256) void conv_stats(
        const float* __restrict__ x,
        const float* __restrict__ cw0, const float* __restrict__ cb0,
        const float* __restrict__ cw1, const float* __restrict__ cb1,
        const float* __restrict__ cw2, const float* __restrict__ cb2,
        const float* __restrict__ cw3, const float* __restrict__ cb3,
        float* __restrict__ stats) {
    __shared__ float red[256];
    int bc = blockIdx.x, tid = threadIdx.x;
    int b = bc >> 8, c = bc & 255, g = c >> 6;
    const float* plane = x + (size_t)bc * NSP;
    const float* w3 = cw0 + c * 9;  const float* w5 = cw1 + c * 25;
    const float* w7 = cw2 + c * 49; const float* w9 = cw3 + c * 81;
    float s[4] = {0, 0, 0, 0}, s2[4] = {0, 0, 0, 0};
    for (int p = tid; p < NSP; p += 256) {
        float y[4];
        conv_all(plane, p >> 6, p & 63, w3, w5, w7, w9, cb0[c], cb1[c], cb2[c], cb3[c], y);
        #pragma unroll
        for (int i = 0; i < 4; ++i) { s[i] += y[i]; s2[i] += y[i] * y[i]; }
    }
    #pragma unroll
    for (int i = 0; i < 4; ++i) {
        red[tid] = s[i]; __syncthreads();
        for (int o = 128; o; o >>= 1) { if (tid < o) red[tid] += red[tid + o]; __syncthreads(); }
        if (tid == 0) atomicAdd(&stats[((i * NB + b) * NGROUPS + g) * 2 + 0], red[0]);
        __syncthreads();
        red[tid] = s2[i]; __syncthreads();
        for (int o = 128; o; o >>= 1) { if (tid < o) red[tid] += red[tid + o]; __syncthreads(); }
        if (tid == 0) atomicAdd(&stats[((i * NB + b) * NGROUPS + g) * 2 + 1], red[0]);
        __syncthreads();
    }
}

// Pass 2: recompute convs, GroupNorm + sigmoid + softmax fusion -> fused fp32 NCHW + xs bf16 [b,n,c]
__global__ __launch_bounds__(256) void conv_fuse(
        const float* __restrict__ x,
        const float* __restrict__ cw0, const float* __restrict__ cb0,
        const float* __restrict__ cw1, const float* __restrict__ cb1,
        const float* __restrict__ cw2, const float* __restrict__ cb2,
        const float* __restrict__ cw3, const float* __restrict__ cb3,
        const float* __restrict__ gnw, const float* __restrict__ gnb,
        const float* __restrict__ fwts, const float* __restrict__ stats,
        float* __restrict__ fused, bf16* __restrict__ xs) {
    int bc = blockIdx.x, tid = threadIdx.x;
    int b = bc >> 8, c = bc & 255, g = c >> 6;
    const float* plane = x + (size_t)bc * NSP;
    const float* w3 = cw0 + c * 9;  const float* w5 = cw1 + c * 25;
    const float* w7 = cw2 + c * 49; const float* w9 = cw3 + c * 81;
    // softmax of the 4 fusion scalars
    float f0 = fwts[0], f1 = fwts[1], f2 = fwts[2], f3 = fwts[3];
    float mx = fmaxf(fmaxf(f0, f1), fmaxf(f2, f3));
    float e0 = __expf(f0 - mx), e1 = __expf(f1 - mx), e2 = __expf(f2 - mx), e3 = __expf(f3 - mx);
    float inv = 1.0f / (e0 + e1 + e2 + e3);
    float wt[4] = {e0 * inv, e1 * inv, e2 * inv, e3 * inv};
    float mean[4], rstd[4], gw[4], gb[4];
    #pragma unroll
    for (int i = 0; i < 4; ++i) {
        float s  = stats[((i * NB + b) * NGROUPS + g) * 2 + 0];
        float s2 = stats[((i * NB + b) * NGROUPS + g) * 2 + 1];
        mean[i] = s / GN_COUNT;
        float var = s2 / GN_COUNT - mean[i] * mean[i];
        rstd[i] = rsqrtf(var + 1e-5f);
        gw[i] = gnw[i * NC + c]; gb[i] = gnb[i * NC + c];
    }
    for (int p = tid; p < NSP; p += 256) {
        float y[4];
        conv_all(plane, p >> 6, p & 63, w3, w5, w7, w9, cb0[c], cb1[c], cb2[c], cb3[c], y);
        float acc = 0.0f;
        #pragma unroll
        for (int i = 0; i < 4; ++i) {
            float yn = (y[i] - mean[i]) * rstd[i] * gw[i] + gb[i];
            acc += wt[i] / (1.0f + __expf(-yn));
        }
        fused[(size_t)bc * NSP + p] = acc;
        xs[((size_t)b * NSP + p) * NC + c] = (bf16)acc;   // NHWC token layout (== v)
    }
}

// ---------------- WMMA kernels ----------------
// QK projection: [65536,256](bf16) x [256,512](bf16,row=output-chan) -> elu+1 -> q/k bf16 (+ k colsum)
__global__ __launch_bounds__(256) void qk_gemm(
        const bf16* __restrict__ xs, const bf16* __restrict__ wt,
        bf16* __restrict__ qbuf, bf16* __restrict__ kbuf, float* __restrict__ ksum) {
    int wave = threadIdx.x >> 5, lane = threadIdx.x & 31;
    int half = lane >> 4, l16 = lane & 15;
    int m0 = blockIdx.x * 16;
    int ntile = blockIdx.y * 8 + wave;        // 0..31
    int ncol = ntile * 16 + l16;              // 0..511
    const bf16* arow = xs + (size_t)(m0 + l16) * NC;
    const bf16* brow = wt + (size_t)ncol * NC;
    floatx8 acc = {0, 0, 0, 0, 0, 0, 0, 0};
    #pragma unroll
    for (int k0 = 0; k0 < 256; k0 += 32) {
        bf16x8 alo = *(const bf16x8*)(arow + k0 + half * 8);
        bf16x8 ahi = *(const bf16x8*)(arow + k0 + 16 + half * 8);
        bf16x8 blo = *(const bf16x8*)(brow + k0 + half * 8);
        bf16x8 bhi = *(const bf16x8*)(brow + k0 + 16 + half * 8);
        bf16x16 a, bb;
        #pragma unroll
        for (int i = 0; i < 8; ++i) { a[i] = alo[i]; a[i + 8] = ahi[i]; bb[i] = blo[i]; bb[i + 8] = bhi[i]; }
        acc = __builtin_amdgcn_wmma_f32_16x16x32_bf16(false, a, false, bb, (short)0, acc, false, false);
    }
    int b = m0 >> 12;
    float klocal = 0.0f;
    #pragma unroll
    for (int r = 0; r < 8; ++r) {
        int m = m0 + r + half * 8;
        float v = acc[r];
        float e = v > 0.0f ? v + 1.0f : __expf(v);   // elu(v)+1
        if (ncol < 256) {
            qbuf[(size_t)m * NC + ncol] = (bf16)e;
        } else {
            kbuf[(size_t)m * NC + (ncol - 256)] = (bf16)e;
            klocal += e;
        }
    }
    if (ncol >= 256) atomicAdd(&ksum[b * NC + (ncol - 256)], klocal);
}

// RoPE on q and k (angle indexed by the w coordinate, as in the reference broadcast)
__global__ void rope_qk(const bf16* __restrict__ qbuf, const bf16* __restrict__ kbuf,
                        bf16* __restrict__ qrope, bf16* __restrict__ krope) {
    int idx = blockIdx.x * blockDim.x + threadIdx.x;   // (m, pair j)
    if (idx >= NM * 128) return;
    int m = idx >> 7, j = idx & 127;
    int wc = (m & (NSP - 1)) & 63;
    float theta = __powf(10000.0f, -(float)j * (1.0f / 128.0f));
    float s, c;
    __sincosf((float)wc * theta, &s, &c);
    size_t base = (size_t)m * NC + 2 * j;
    float q0 = (float)qbuf[base], q1 = (float)qbuf[base + 1];
    qrope[base] = (bf16)(c * q0 - s * q1); qrope[base + 1] = (bf16)(s * q0 + c * q1);
    float k0 = (float)kbuf[base], k1 = (float)kbuf[base + 1];
    krope[base] = (bf16)(c * k0 - s * k1); krope[base + 1] = (bf16)(s * k0 + c * k1);
}

// z[m,h] = 1 / (q[m, head] . k_mean[head] + 1e-6)
__global__ void z_calc(const bf16* __restrict__ qbuf, const float* __restrict__ ksum,
                       float* __restrict__ z) {
    int idx = blockIdx.x * blockDim.x + threadIdx.x;
    if (idx >= NM * NHEADS) return;
    int m = idx >> 2, h = idx & 3, b = m >> 12;
    const bf16* q = qbuf + (size_t)m * NC + h * NHD;
    const float* ks = ksum + b * NC + h * NHD;
    float acc = 0.0f;
    #pragma unroll 8
    for (int d = 0; d < NHD; ++d) acc += (float)q[d] * ks[d];
    z[idx] = 1.0f / (acc * (1.0f / 4096.0f) + 1e-6f);
}

// kv[b,h] = (1/n) k_rope^T v : 64x64 per (b,head).
// Tiles staged memory->LDS with CDNA5 GLOBAL_LOAD_ASYNC_TO_LDS_B128 (ASYNCcnt),
// then consumed by v_wmma. K split over blocks, fp32 atomic accumulate.
__global__ __launch_bounds__(512) void kv_gemm(const bf16* __restrict__ krope,
                                               const bf16* __restrict__ xs,
                                               float* __restrict__ kv) {
    // rows padded to 80 bf16 (160 B): keeps every 16 B async store naturally aligned
    __shared__ __align__(16) bf16 kt[32][80];
    __shared__ __align__(16) bf16 vt[32][80];
    int bh = blockIdx.x, b = bh >> 2, h = bh & 3;
    int tid = threadIdx.x;
    int wave = tid >> 5, lane = tid & 31, half = lane >> 4, l16 = lane & 15;
    int dt = wave >> 2, et = wave & 3;          // 4x4 grid of 16x16 tiles over [64,64]
    // async staging assignment: threads 0..255 -> kt, 256..511 -> vt; 16 B per thread
    int sel = tid >> 8;
    int t8  = tid & 255;
    int snn = t8 >> 3;                          // 0..31 (row)
    int scol = (t8 & 7) * 8;                    // 0..56 (bf16 col, 16 B chunks)
    floatx8 acc = {0, 0, 0, 0, 0, 0, 0, 0};
    int kbase = blockIdx.y * 512;
    for (int ks = kbase; ks < kbase + 512; ks += 32) {
        size_t gidx = ((size_t)(b * NSP + ks + snn)) * NC + h * NHD + scol;
        const bf16* gp = (sel ? xs : krope) + gidx;
        bf16* lp = sel ? &vt[snn][scol] : &kt[snn][scol];
        async_load_b128_to_lds(gp, lp);      // memory -> LDS, no VGPR round trip
        wait_asynccnt0();                    // wave's own async ops landed
        __syncthreads();                     // all waves' staging visible
        int d = dt * 16 + l16, e = et * 16 + l16;
        bf16x16 a, bb;
        #pragma unroll
        for (int i = 0; i < 8; ++i) {
            a[i]     = kt[half * 16 + i][d];         // A: K = half*16 + e  (e<8)
            a[i + 8] = kt[half * 16 + 8 + i][d];     //    K = half*16 + e  (e>=8)
        }
        #pragma unroll
        for (int i = 0; i < 16; ++i) bb[i] = vt[half * 16 + i][e];   // B: K = half*16 + e
        acc = __builtin_amdgcn_wmma_f32_16x16x32_bf16(false, a, false, bb, (short)0, acc, false, false);
        __syncthreads();
    }
    #pragma unroll
    for (int r = 0; r < 8; ++r) {
        int d = dt * 16 + r + half * 8, e = et * 16 + l16;
        atomicAdd(&kv[((size_t)bh * NHD + d) * NHD + e], acc[r] * (1.0f / 4096.0f));
    }
}

// attn = (q_rope @ kv) * z -> xo fp32 [b,n,c]
__global__ __launch_bounds__(128) void attn_gemm(const bf16* __restrict__ qrope,
                                                 const float* __restrict__ kv,
                                                 const float* __restrict__ z,
                                                 float* __restrict__ xo) {
    int wave = threadIdx.x >> 5, lane = threadIdx.x & 31;
    int half = lane >> 4, l16 = lane & 15;
    int m0 = blockIdx.x * 16, h = blockIdx.y;
    int b = m0 >> 12, bh = b * NHEADS + h;
    int ecol = wave * 16 + l16;
    const bf16* arow = qrope + (size_t)(m0 + l16) * NC + h * NHD;
    floatx8 acc = {0, 0, 0, 0, 0, 0, 0, 0};
    #pragma unroll
    for (int k0 = 0; k0 < 64; k0 += 32) {
        bf16x8 alo = *(const bf16x8*)(arow + k0 + half * 16);
        bf16x8 ahi = *(const bf16x8*)(arow + k0 + half * 16 + 8);
        bf16x16 a, bb;
        #pragma unroll
        for (int i = 0; i < 8; ++i) { a[i] = alo[i]; a[i + 8] = ahi[i]; }
        #pragma unroll
        for (int i = 0; i < 16; ++i)
            bb[i] = (bf16)kv[((size_t)bh * NHD + k0 + half * 16 + i) * NHD + ecol];
        acc = __builtin_amdgcn_wmma_f32_16x16x32_bf16(false, a, false, bb, (short)0, acc, false, false);
    }
    #pragma unroll
    for (int r = 0; r < 8; ++r) {
        int m = m0 + r + half * 8;
        xo[(size_t)m * NC + h * NHD + ecol] = acc[r] * z[(size_t)m * NHEADS + h];
    }
}

// LePE 3x3 depthwise on fused (NCHW) + transpose-add of xo -> out [b,c,h,w]
__global__ __launch_bounds__(256) void lepe_out(const float* __restrict__ fused,
                                                const float* __restrict__ xo,
                                                const float* __restrict__ lw,
                                                const float* __restrict__ lb,
                                                float* __restrict__ out) {
    int bc = blockIdx.x, tid = threadIdx.x;
    int b = bc >> 8, c = bc & 255;
    const float* plane = fused + (size_t)bc * NSP;
    const float* w = lw + c * 9;
    float bias = lb[c];
    for (int p = tid; p < NSP; p += 256) {
        int h = p >> 6, wc = p & 63;
        float acc = bias;
        #pragma unroll
        for (int dh = -1; dh <= 1; ++dh) {
            int hh = h + dh;
            if (hh < 0 || hh >= NHW) continue;
            #pragma unroll
            for (int dw = -1; dw <= 1; ++dw) {
                int ww = wc + dw;
                if (ww < 0 || ww >= NHW) continue;
                acc += plane[hh * NHW + ww] * w[(dh + 1) * 3 + (dw + 1)];
            }
        }
        out[(size_t)bc * NSP + p] = xo[((size_t)b * NSP + p) * NC + c] + acc;
    }
}

// ---------------- launcher ----------------
extern "C" void kernel_launch(void* const* d_in, const int* in_sizes, int n_in,
                              void* d_out, int out_size, void* d_ws, size_t ws_size,
                              hipStream_t stream) {
    const float* x    = (const float*)d_in[0];
    const float* cw0  = (const float*)d_in[1];
    const float* cb0  = (const float*)d_in[2];
    const float* cw1  = (const float*)d_in[3];
    const float* cb1  = (const float*)d_in[4];
    const float* cw2  = (const float*)d_in[5];
    const float* cb2  = (const float*)d_in[6];
    const float* cw3  = (const float*)d_in[7];
    const float* cb3  = (const float*)d_in[8];
    const float* gnw  = (const float*)d_in[9];
    const float* gnb  = (const float*)d_in[10];
    const float* fwts = (const float*)d_in[11];
    const float* qkw  = (const float*)d_in[12];
    const float* lw   = (const float*)d_in[13];
    const float* lb   = (const float*)d_in[14];

    char* ws = (char*)d_ws;
    float* stats = (float*)(ws + OFF_STATS);
    float* ksum  = (float*)(ws + OFF_KSUM);
    float* kv    = (float*)(ws + OFF_KV);
    float* z     = (float*)(ws + OFF_Z);
    bf16*  qkwb  = (bf16*)(ws + OFF_QKWB);
    float* fused = (float*)(ws + OFF_FUSED);
    bf16*  xs    = (bf16*)(ws + OFF_XS);
    bf16*  qrope = (bf16*)(ws + OFF_QROPE);
    bf16*  krope = (bf16*)(ws + OFF_KROPE);
    bf16*  qbuf  = (bf16*)(ws + OFF_QBUF);
    bf16*  kbuf  = (bf16*)(ws + OFF_KBUF);
    float* xo    = (float*)(ws + OFF_XO);     // aliases qbuf/kbuf (dead by then)

    // 1) zero GN stats, k-sums, kv accumulators
    zero_f32<<<((int)ZERO_FLOATS + 255) / 256, 256, 0, stream>>>((float*)ws, (int)ZERO_FLOATS);
    // 2) qk weight -> bf16
    cvt_bf16<<<(512 * 256 + 255) / 256, 256, 0, stream>>>(qkw, qkwb, 512 * 256);
    // 3) conv + GN statistics
    conv_stats<<<NB * NC, 256, 0, stream>>>(x, cw0, cb0, cw1, cb1, cw2, cb2, cw3, cb3, stats);
    // 4) conv recompute + GN + sigmoid + softmax fusion
    conv_fuse<<<NB * NC, 256, 0, stream>>>(x, cw0, cb0, cw1, cb1, cw2, cb2, cw3, cb3,
                                           gnw, gnb, fwts, stats, fused, xs);
    // 5) QK projection GEMM (WMMA bf16) with elu+1 epilogue + k column sums
    qk_gemm<<<dim3(NM / 16, 4), 256, 0, stream>>>(xs, qkwb, qbuf, kbuf, ksum);
    // 6) RoPE
    rope_qk<<<(NM * 128 + 255) / 256, 256, 0, stream>>>(qbuf, kbuf, qrope, krope);
    // 7) z normalizer
    z_calc<<<(NM * NHEADS + 255) / 256, 256, 0, stream>>>(qbuf, ksum, z);
    // 8) kv = k_rope^T v / n  (WMMA, async-to-LDS staged, K-chunked)
    kv_gemm<<<dim3(NB * NHEADS, 8), 512, 0, stream>>>(krope, xs, kv);
    // 9) attn = (q_rope kv) * z  (WMMA)
    attn_gemm<<<dim3(NM / 16, NHEADS), 128, 0, stream>>>(qrope, kv, z, xo);
    // 10) LePE + transpose to NCHW output
    lepe_out<<<NB * NC, 256, 0, stream>>>(fused, xo, lw, lb, (float*)d_out);

    (void)in_sizes; (void)n_in; (void)out_size; (void)ws_size;
}
